// ScaledDotProductAttention_15006615733239
// MI455X (gfx1250) — matrix operand order repaired and checked by reading, not compiled, not run on hardware
//
#include <hip/hip_runtime.h>
#include <hip/hip_bf16.h>

// ---------------------------------------------------------------------------
// MI455X (gfx1250) multi-head attention
//   - all GEMMs via v_wmma_f32_16x16x32_bf16 (wave32), 32x64 tile per wave
//   - V kept transposed [B,H,Dh,S] so P.V B-fragments are contiguous loads
//   - 193 KB LDS per workgroup in attention (128KB f32 scores + 64KB bf16 P)
//   - softmax normalization folded into the output store (no 3rd LDS sweep)
// ---------------------------------------------------------------------------

typedef __attribute__((ext_vector_type(16))) __bf16 v16bf;
typedef __attribute__((ext_vector_type(8)))  __bf16 v8bf;
typedef __attribute__((ext_vector_type(8)))  float  v8f;

#define EMBED 1024
#define HEADS 16
#define HDIM  64
#define BATCH 4
#define SEQ   2048
#define NTOK  (BATCH * SEQ)             // 8192 rows
static constexpr float SCALE = 0.125f;  // 1/sqrt(64)

// ---- WMMA fragment helpers (layouts per CDNA5 ISA 7.12.2, wave32) ----------

// A (16x32 bf16): lane&15 = row M; lane<16 holds K {k0..k0+7, k0+16..k0+23},
// lane>=16 holds K {k0+8..k0+15, k0+24..k0+31}.  Source fp32 row-major.
static __device__ inline v16bf frag_a_f32(const float* __restrict__ base,
                                          int ld, int m0, int k0) {
  const int lane = threadIdx.x & 31;
  const int m  = m0 + (lane & 15);
  const int kb = k0 + ((lane >> 4) << 3);
  const float* p = base + (size_t)m * ld + kb;
  v16bf a;
#pragma unroll
  for (int j = 0; j < 8; ++j) {
    a[j]     = (__bf16)p[j];
    a[j + 8] = (__bf16)p[j + 16];
  }
  return a;
}

// Same A layout, bf16 source (global or LDS).
static __device__ inline v16bf frag_a_bf16(const __bf16* base,
                                           int ld, int m0, int k0) {
  const int lane = threadIdx.x & 31;
  const int m  = m0 + (lane & 15);
  const int kb = k0 + ((lane >> 4) << 3);
  const __bf16* p = base + (size_t)m * ld + kb;
  v8bf lo = *(const v8bf*)(p);
  v8bf hi = *(const v8bf*)(p + 16);
  return __builtin_shufflevector(lo, hi, 0, 1, 2, 3, 4, 5, 6, 7,
                                 8, 9, 10, 11, 12, 13, 14, 15);
}

// B (32x16 bf16), stored as stored[n][k] with k contiguous:
// lane&15 = column N, lane<16 holds K {k0..k0+7, k0+16..k0+23},
// lane>=16 holds K {k0+8..k0+15, k0+24..k0+31}.
static __device__ inline v16bf frag_b_rowk(const __bf16* __restrict__ base,
                                           int ld, int n0, int k0) {
  const int lane = threadIdx.x & 31;
  const int n  = n0 + (lane & 15);
  const int kb = k0 + ((lane >> 4) << 3);
  const __bf16* p = base + (size_t)n * ld + kb;
  v8bf lo = *(const v8bf*)(p);
  v8bf hi = *(const v8bf*)(p + 16);
  return __builtin_shufflevector(lo, hi, 0, 1, 2, 3, 4, 5, 6, 7,
                                 8, 9, 10, 11, 12, 13, 14, 15);
}

static __device__ inline v8f wmma_bf16(v16bf a, v16bf b, v8f c) {
  return __builtin_amdgcn_wmma_f32_16x16x32_bf16(
      false, a, false, b, (short)0, c, false, false);
}

// ---------------------------------------------------------------------------
// Kernel 1: fp32 -> bf16 convert (weights)
// ---------------------------------------------------------------------------
__global__ void cvt_f32_bf16_kernel(const float* __restrict__ src,
                                    __bf16* __restrict__ dst, int n) {
  int i = blockIdx.x * blockDim.x + threadIdx.x;
  if (i < n) dst[i] = (__bf16)src[i];
}

// ---------------------------------------------------------------------------
// Kernel 2: input projections.  Each wave owns a 32x64 output tile
// (2 A-frags x 4 B-frags -> 8 WMMAs per K-step).
//   VTRANS=false: dst[b,h,s,d]   (Q and K layout)
//   VTRANS=true : dst[b,h,d,s]   (V transposed for contiguous P.V fragments)
// ---------------------------------------------------------------------------
template <bool VTRANS>
__global__ void proj_kernel(const float* __restrict__ X,
                            const __bf16* __restrict__ W,
                            const float* __restrict__ bias,
                            __bf16* __restrict__ dst) {
  const int lane = threadIdx.x & 31;
  const int wave = threadIdx.x >> 5;
  const int tile = blockIdx.x * 8 + wave;   // 256 (M) x 16 (N) wave tiles
  const int n0 = (tile & 15) << 6;          // 64-wide N tile
  const int m0 = (tile >> 4) << 5;          // 32-tall M tile

  v8f c[2][4] = {};
  for (int k0 = 0; k0 < EMBED; k0 += 32) {
    __builtin_prefetch(X + (size_t)(m0 + (lane & 15)) * EMBED + k0 + 64, 0, 1);
    v16bf a0 = frag_a_f32(X, EMBED, m0,      k0);
    v16bf a1 = frag_a_f32(X, EMBED, m0 + 16, k0);
    v16bf b0 = frag_b_rowk(W, EMBED, n0,      k0);
    v16bf b1 = frag_b_rowk(W, EMBED, n0 + 16, k0);
    v16bf b2 = frag_b_rowk(W, EMBED, n0 + 32, k0);
    v16bf b3 = frag_b_rowk(W, EMBED, n0 + 48, k0);
    c[0][0] = wmma_bf16(a0, b0, c[0][0]);
    c[0][1] = wmma_bf16(a0, b1, c[0][1]);
    c[0][2] = wmma_bf16(a0, b2, c[0][2]);
    c[0][3] = wmma_bf16(a0, b3, c[0][3]);
    c[1][0] = wmma_bf16(a1, b0, c[1][0]);
    c[1][1] = wmma_bf16(a1, b1, c[1][1]);
    c[1][2] = wmma_bf16(a1, b2, c[1][2]);
    c[1][3] = wmma_bf16(a1, b3, c[1][3]);
  }

#pragma unroll
  for (int j = 0; j < 4; ++j) {
    const int o  = n0 + (j << 4) + (lane & 15);
    const float bv = bias[o];
    const int h = o >> 6, d = o & 63;
#pragma unroll
    for (int i = 0; i < 2; ++i) {
#pragma unroll
      for (int r = 0; r < 8; ++r) {
        const int m  = m0 + (i << 4) + r + ((lane >> 4) << 3);
        const int bb = m >> 11;          // / SEQ
        const int s  = m & (SEQ - 1);
        size_t idx;
        if (VTRANS)
          idx = (((size_t)(bb * HEADS + h)) * HDIM + d) * SEQ + s;
        else
          idx = (((size_t)(bb * HEADS + h)) * SEQ + s) * HDIM + d;
        dst[idx] = (__bf16)(c[i][j][r] + bv);
      }
    }
  }
}

// ---------------------------------------------------------------------------
// Kernel 3: attention for one (b,h,16-query tile).
// LDS: 16x2048 f32 scores (128KB) + 16x2048 bf16 P (64KB) + reductions.
// ---------------------------------------------------------------------------
__global__ void attn_kernel(const __bf16* __restrict__ Q,
                            const __bf16* __restrict__ K,
                            const __bf16* __restrict__ Vt,
                            __bf16* __restrict__ ctx) {
  extern __shared__ float smem[];
  float*  sc    = smem;                         // [16][2048] f32 scores
  float*  red   = smem + 16 * SEQ;              // [16][16]
  float*  rstat = red + 256;                    // [16]
  __bf16* pb    = (__bf16*)(rstat + 16);        // [16][2048] bf16 exp(P)

  const int lane = threadIdx.x & 31;
  const int wave = threadIdx.x >> 5;
  const int bh = blockIdx.x >> 7;               // / (SEQ/16)
  const int q0 = (blockIdx.x & 127) << 4;
  const __bf16* Qb  = Q  + (size_t)bh * SEQ * HDIM;
  const __bf16* Kb  = K  + (size_t)bh * SEQ * HDIM;
  const __bf16* Vtb = Vt + (size_t)bh * HDIM * SEQ;

  // ---- Step 1: scores = scale * Q Kt ; waves stripe over 128 key tiles ----
  v16bf qa0 = frag_a_bf16(Qb, HDIM, q0, 0);
  v16bf qa1 = frag_a_bf16(Qb, HDIM, q0, 32);
  for (int kt = wave; kt < SEQ / 16; kt += 8) {
    const int n0 = kt << 4;
    v16bf b0 = frag_b_rowk(Kb, HDIM, n0, 0);
    v16bf b1 = frag_b_rowk(Kb, HDIM, n0, 32);
    v8f c = {};
    c = wmma_bf16(qa0, b0, c);
    c = wmma_bf16(qa1, b1, c);
    const int col = n0 + (lane & 15);
#pragma unroll
    for (int r = 0; r < 8; ++r) {
      const int row = r + ((lane >> 4) << 3);
      sc[row * SEQ + col] = c[r] * SCALE;
    }
  }
  __syncthreads();

  // ---- Step 2: row max, then exp -> bf16 P strip + row sums ---------------
  {
    const int row = threadIdx.x & 15;
    const int chunk = threadIdx.x >> 4;         // 0..15, 128 cols each
    float* srow = sc + row * SEQ;
    __bf16* prow = pb + row * SEQ;
    const int c0 = chunk << 7;

    float mx = -3.4e38f;
    for (int j = c0; j < c0 + 128; ++j) mx = fmaxf(mx, srow[j]);
    red[row * 16 + chunk] = mx;
    __syncthreads();
    if (chunk == 0) {
      float m = red[row * 16];
      for (int j = 1; j < 16; ++j) m = fmaxf(m, red[row * 16 + j]);
      rstat[row] = m;
    }
    __syncthreads();

    const float rm = rstat[row];
    float sum = 0.f;
    for (int j = c0; j < c0 + 128; ++j) {
      float e = __expf(srow[j] - rm);           // e <= 1, safe in bf16
      prow[j] = (__bf16)e;
      sum += e;
    }
    red[row * 16 + chunk] = sum;
    __syncthreads();
    if (chunk == 0) {
      float t = 0.f;
      for (int j = 0; j < 16; ++j) t += red[row * 16 + j];
      rstat[row] = 1.0f / t;                    // folded into output store
    }
  }
  __syncthreads();

  // ---- Step 3: O = P V.  wave&3 picks d-tile, wave>>2 picks key half. -----
  const int nt = wave & 3;
  const int half = wave >> 2;
  const int n0 = nt << 4;
  v8f c = {};
  for (int kk = half * 1024; kk < half * 1024 + 1024; kk += 32) {
    v16bf a = frag_a_bf16(pb, SEQ, 0, kk);               // ds_load_b128 x2
    v16bf b = frag_b_rowk(Vtb, SEQ, n0, kk);             // 2 x 16B
    c = wmma_bf16(a, b, c);
  }

  // Cross-wave (key-half) reduction through the now-free score strip.
  if (wave >= 4) {
    float* ex = sc + (wave - 4) * 256;
#pragma unroll
    for (int r = 0; r < 8; ++r) ex[lane * 8 + r] = c[r];
  }
  __syncthreads();
  if (wave < 4) {
    const float* ex = sc + wave * 256;
    const int h = bh & (HEADS - 1);
    const int b = bh >> 4;
    const int ocol = h * HDIM + n0 + (lane & 15);
#pragma unroll
    for (int r = 0; r < 8; ++r) {
      const int rq = r + ((lane >> 4) << 3);
      const float inv = rstat[rq];
      const float v = (c[r] + ex[lane * 8 + r]) * inv;
      ctx[((size_t)(b * SEQ + q0 + rq)) * EMBED + ocol] = (__bf16)v;
    }
  }
}

// ---------------------------------------------------------------------------
// Kernel 4: output projection (32x64 wave tile).  fp32 out = ctx.Wo^T + bo
// ---------------------------------------------------------------------------
__global__ void out_proj_kernel(const __bf16* __restrict__ A,
                                const __bf16* __restrict__ W,
                                const float* __restrict__ bias,
                                float* __restrict__ out) {
  const int lane = threadIdx.x & 31;
  const int wave = threadIdx.x >> 5;
  const int tile = blockIdx.x * 8 + wave;
  const int n0 = (tile & 15) << 6;
  const int m0 = (tile >> 4) << 5;

  v8f c[2][4] = {};
  for (int k0 = 0; k0 < EMBED; k0 += 32) {
    __builtin_prefetch(A + (size_t)(m0 + (lane & 15)) * EMBED + k0 + 64, 0, 1);
    v16bf a0 = frag_a_bf16(A, EMBED, m0,      k0);
    v16bf a1 = frag_a_bf16(A, EMBED, m0 + 16, k0);
    v16bf b0 = frag_b_rowk(W, EMBED, n0,      k0);
    v16bf b1 = frag_b_rowk(W, EMBED, n0 + 16, k0);
    v16bf b2 = frag_b_rowk(W, EMBED, n0 + 32, k0);
    v16bf b3 = frag_b_rowk(W, EMBED, n0 + 48, k0);
    c[0][0] = wmma_bf16(a0, b0, c[0][0]);
    c[0][1] = wmma_bf16(a0, b1, c[0][1]);
    c[0][2] = wmma_bf16(a0, b2, c[0][2]);
    c[0][3] = wmma_bf16(a0, b3, c[0][3]);
    c[1][0] = wmma_bf16(a1, b0, c[1][0]);
    c[1][1] = wmma_bf16(a1, b1, c[1][1]);
    c[1][2] = wmma_bf16(a1, b2, c[1][2]);
    c[1][3] = wmma_bf16(a1, b3, c[1][3]);
  }

#pragma unroll
  for (int j = 0; j < 4; ++j) {
    const int o = n0 + (j << 4) + (lane & 15);
    const float bv = bias[o];
#pragma unroll
    for (int i = 0; i < 2; ++i) {
#pragma unroll
      for (int r = 0; r < 8; ++r) {
        const int m = m0 + (i << 4) + r + ((lane >> 4) << 3);
        out[(size_t)m * EMBED + o] = c[i][j][r] + bv;
      }
    }
  }
}

// ---------------------------------------------------------------------------
// Host launcher
// ---------------------------------------------------------------------------
extern "C" void kernel_launch(void* const* d_in, const int* in_sizes, int n_in,
                              void* d_out, int out_size, void* d_ws, size_t ws_size,
                              hipStream_t stream) {
  (void)in_sizes; (void)n_in; (void)out_size; (void)ws_size;

  const float* query = (const float*)d_in[0];
  const float* keyi  = (const float*)d_in[1];
  const float* value = (const float*)d_in[2];
  const float* Wq = (const float*)d_in[3];
  const float* bq = (const float*)d_in[4];
  const float* Wk = (const float*)d_in[5];
  const float* bk = (const float*)d_in[6];
  const float* Wv = (const float*)d_in[7];
  const float* bv = (const float*)d_in[8];
  const float* Wo = (const float*)d_in[9];
  const float* bo = (const float*)d_in[10];
  float* out = (float*)d_out;

  char* ws = (char*)d_ws;
  const size_t WBYTES = (size_t)EMBED * EMBED * sizeof(__bf16);   // 2 MB
  const size_t ABYTES = (size_t)NTOK * EMBED * sizeof(__bf16);    // 16 MB
  __bf16* Wqb = (__bf16*)(ws);
  __bf16* Wkb = (__bf16*)(ws + 1 * WBYTES);
  __bf16* Wvb = (__bf16*)(ws + 2 * WBYTES);
  __bf16* Wob = (__bf16*)(ws + 3 * WBYTES);
  __bf16* Qb  = (__bf16*)(ws + 4 * WBYTES);
  __bf16* Kb  = (__bf16*)(ws + 4 * WBYTES + 1 * ABYTES);
  __bf16* Vtb = (__bf16*)(ws + 4 * WBYTES + 2 * ABYTES);
  __bf16* Cx  = (__bf16*)(ws + 4 * WBYTES + 3 * ABYTES);

  const int WN = EMBED * EMBED;
  const dim3 blk(256);

  // 1. Convert weights to bf16
  cvt_f32_bf16_kernel<<<dim3(WN / 256), blk, 0, stream>>>(Wq, Wqb, WN);
  cvt_f32_bf16_kernel<<<dim3(WN / 256), blk, 0, stream>>>(Wk, Wkb, WN);
  cvt_f32_bf16_kernel<<<dim3(WN / 256), blk, 0, stream>>>(Wv, Wvb, WN);
  cvt_f32_bf16_kernel<<<dim3(WN / 256), blk, 0, stream>>>(Wo, Wob, WN);

  // 2. Projections: (8192/32) x (1024/64) = 4096 wave tiles -> 512 blocks
  const dim3 gproj((NTOK / 32) * (EMBED / 64) / 8);
  proj_kernel<false><<<gproj, blk, 0, stream>>>(query, Wqb, bq, Qb);
  proj_kernel<false><<<gproj, blk, 0, stream>>>(keyi,  Wkb, bk, Kb);
  proj_kernel<true ><<<gproj, blk, 0, stream>>>(value, Wvb, bv, Vtb);

  // 3. Attention: B*H*(S/16) = 8192 workgroups, ~193 KB dynamic LDS each
  const size_t smem = (16 * SEQ + 256 + 16) * sizeof(float)
                    + (16 * SEQ) * sizeof(__bf16);
  attn_kernel<<<dim3(BATCH * HEADS * (SEQ / 16)), blk, smem, stream>>>(Qb, Kb, Vtb, Cx);

  // 4. Output projection -> fp32 d_out
  out_proj_kernel<<<gproj, blk, 0, stream>>>(Cx, Wob, bo, out);
}
